// DeepseekV3Attention_40492951666928
// MI455X (gfx1250) — matrix operand order repaired and checked by reading, not compiled, not run on hardware
//
#include <hip/hip_runtime.h>

// ---------------- constants (match reference) ----------------
#define Bz   1
#define Sz   2048
#define HIDz 2048
#define NHz  16
#define DNz  128
#define DRz  64
#define DHz  192       // DN + DR
#define DVz  128
#define QLz  1536
#define KLz  512
#define EPSz 1e-6f
#define SCALING_F 0.13523378f   // DH^-0.5 * (0.1*ln40+1)^2

typedef __bf16 bf16;
typedef __attribute__((ext_vector_type(16))) bf16  v16bf;
typedef __attribute__((ext_vector_type(8)))  float v8f;
typedef __attribute__((ext_vector_type(4)))  int   v4i;

__device__ __forceinline__ v8f vzero8() {
  v8f z = {0.f,0.f,0.f,0.f,0.f,0.f,0.f,0.f};
  return z;
}

// A-fragment (16x32 bf16, M x K): lane r holds row r; K chunks at
// k + 8*(lane>>4) (VGPRs 0-3) and k + 16 + 8*(lane>>4) (VGPRs 4-7).
__device__ __forceinline__ v16bf load_a_frag(const bf16* base, int ld, int lane) {
  const int r    = lane & 15;
  const int half = (lane >> 4) & 1;
  const bf16* p  = base + (size_t)r * ld + half * 8;
  v16bf f;
  reinterpret_cast<v4i*>(&f)[0] = *reinterpret_cast<const v4i*>(p);
  reinterpret_cast<v4i*>(&f)[1] = *reinterpret_cast<const v4i*>(p + 16);
  return f;
}

// B-fragment (32x16 bf16, K x N) built from W stored (N x K) row-major:
// lane n holds W[n0+n, k + 8*(lane>>4) .. +7] and W[n0+n, k + 16 + 8*(lane>>4) .. +7].
__device__ __forceinline__ v16bf load_b_frag(const bf16* base, int ld, int lane) {
  const int n    = lane & 15;
  const int koff = ((lane >> 4) & 1) * 8;
  const bf16* p  = base + (size_t)n * ld + koff;
  v16bf f;
  reinterpret_cast<v4i*>(&f)[0] = *reinterpret_cast<const v4i*>(p);
  reinterpret_cast<v4i*>(&f)[1] = *reinterpret_cast<const v4i*>(p + 16);
  return f;
}

__device__ __forceinline__ v8f wmma_bf16(v16bf a, v16bf b, v8f c) {
  return __builtin_amdgcn_wmma_f32_16x16x32_bf16(
      /*neg_a=*/false, a, /*neg_b=*/false, b,
      /*c_mod=*/(short)0, c, /*reuse_a=*/false, /*reuse_b=*/false);
}

// ---------------- fp32 -> bf16 convert ----------------
__global__ void f2bf_kernel(const float* __restrict__ x, bf16* __restrict__ y, int n) {
  int i = blockIdx.x * blockDim.x + threadIdx.x;
  int stride = gridDim.x * blockDim.x;
  for (; i < n; i += stride) y[i] = (bf16)x[i];
}

// ---------------- GEMM: C(f32, MxN) = A(bf16 MxK) * W(bf16 NxK)^T ----------
// 256 threads = 8 waves arranged 4(M) x 2(N); wave tile = 16 x 64.
// Ping-pong software pipeline (K-tiles of 64, even tile count required):
// load(t+1) -> wmma(setA) -> load(t+2) -> wmma(setB), no buffer copies.
__global__ __launch_bounds__(256) void gemm_bt_kernel(
    const bf16* __restrict__ A, const bf16* __restrict__ W,
    float* __restrict__ C, int M, int N, int K)
{
  const int lane = threadIdx.x & 31;
  const int wave = threadIdx.x >> 5;
  const int m0 = blockIdx.y * 64  + (wave >> 1) * 16;
  const int n0 = blockIdx.x * 128 + (wave & 1) * 64;
  if (m0 >= M || n0 >= N) return;

  const bf16* Arow = A + (size_t)m0 * K;
  const bf16* Wr[4];
#pragma unroll
  for (int j = 0; j < 4; ++j) Wr[j] = W + (size_t)(n0 + 16 * j) * K;

  v8f acc[4];
#pragma unroll
  for (int j = 0; j < 4; ++j) acc[j] = vzero8();

  v16bf a0, a1, bA0[4], bA1[4];   // buffer set A
  v16bf c0, c1, bB0[4], bB1[4];   // buffer set B

  auto loadA = [&](int kk) {
    __builtin_prefetch(Arow + kk + 128, 0, 1);
    a0 = load_a_frag(Arow + kk, K, lane);
    a1 = load_a_frag(Arow + kk + 32, K, lane);
#pragma unroll
    for (int j = 0; j < 4; ++j) {
      bA0[j] = load_b_frag(Wr[j] + kk, K, lane);
      bA1[j] = load_b_frag(Wr[j] + kk + 32, K, lane);
    }
  };
  auto loadB = [&](int kk) {
    a1 = a1; // no-op
    c0 = load_a_frag(Arow + kk, K, lane);
    c1 = load_a_frag(Arow + kk + 32, K, lane);
#pragma unroll
    for (int j = 0; j < 4; ++j) {
      bB0[j] = load_b_frag(Wr[j] + kk, K, lane);
      bB1[j] = load_b_frag(Wr[j] + kk + 32, K, lane);
    }
  };
  auto mmaA = [&]() {
#pragma unroll
    for (int j = 0; j < 4; ++j) acc[j] = wmma_bf16(a0, bA0[j], acc[j]);
#pragma unroll
    for (int j = 0; j < 4; ++j) acc[j] = wmma_bf16(a1, bA1[j], acc[j]);
  };
  auto mmaB = [&]() {
#pragma unroll
    for (int j = 0; j < 4; ++j) acc[j] = wmma_bf16(c0, bB0[j], acc[j]);
#pragma unroll
    for (int j = 0; j < 4; ++j) acc[j] = wmma_bf16(c1, bB1[j], acc[j]);
  };

  // tile count K/64 is even at every call site (K in {2048, 1536, 512})
  const int T = K >> 6;
  loadA(0);
  int t = 0;
  for (; t + 2 < T; t += 2) {
    loadB((t + 1) << 6);
    mmaA();
    loadA((t + 2) << 6);
    mmaB();
  }
  loadB((t + 1) << 6);
  mmaA();
  mmaB();

  const int col = lane & 15, half = lane >> 4;
#pragma unroll
  for (int j = 0; j < 4; ++j)
#pragma unroll
    for (int v = 0; v < 8; ++v)
      C[(size_t)(m0 + v + 8 * half) * N + (n0 + 16 * j + col)] = acc[j][v];
}

// ---------------- RMSNorm: row-wise over `cols`, f32 in / bf16 out --------
__global__ __launch_bounds__(256) void rmsnorm_kernel(
    const float* __restrict__ X, const float* __restrict__ w,
    bf16* __restrict__ Y, int ldx, int cols)
{
  __shared__ float red[256];
  const int row = blockIdx.x;
  const float* xr = X + (size_t)row * ldx;
  float s = 0.f;
  for (int c = threadIdx.x; c < cols; c += blockDim.x) { float v = xr[c]; s += v * v; }
  red[threadIdx.x] = s;
  __syncthreads();
  for (int off = 128; off > 0; off >>= 1) {
    if ((int)threadIdx.x < off) red[threadIdx.x] += red[threadIdx.x + off];
    __syncthreads();
  }
  const float inv = rsqrtf(red[0] / (float)cols + EPSz);
  for (int c = threadIdx.x; c < cols; c += blockDim.x)
    Y[(size_t)row * cols + c] = (bf16)(xr[c] * inv * w[c]);
}

// ---------------- build qq[h][s][0..191]: pass + deinterleave + RoPE -------
__global__ __launch_bounds__(192) void build_q_kernel(
    const float* __restrict__ qraw, const float* __restrict__ cosb,
    const float* __restrict__ sinb, bf16* __restrict__ qq)
{
  const int s = blockIdx.x, h = blockIdx.y, t = threadIdx.x;
  const float* qr = qraw + (size_t)s * (NHz * DHz) + h * DHz;
  bf16* dst = qq + ((size_t)h * Sz + s) * DHz;
  if (t < DNz) {
    dst[t] = (bf16)qr[t];
  } else {
    const int j = t - DNz;             // 0..63
    const float* rot = qr + DNz;
    const float c  = cosb[(size_t)s * DRz + j];
    const float sn = sinb[(size_t)s * DRz + j];
    float o;
    if (j < 32) o = rot[2 * j] * c - rot[2 * j + 1] * sn;
    else { int jj = j - 32; o = rot[2 * jj + 1] * c + rot[2 * jj] * sn; }
    dst[DNz + j] = (bf16)o;
  }
}

// ------- build kk[h][s][0..191] and vT[h][d][s] from kv-up + k_rot RoPE ----
__global__ __launch_bounds__(128) void build_k_kernel(
    const float* __restrict__ kvraw, const float* __restrict__ ckv,
    const float* __restrict__ cosb, const float* __restrict__ sinb,
    bf16* __restrict__ kk, bf16* __restrict__ vT)
{
  const int s = blockIdx.x, h = blockIdx.y, t = threadIdx.x;
  const float* kvr = kvraw + (size_t)s * (NHz * (DNz + DVz)) + h * (DNz + DVz);
  bf16* kdst = kk + ((size_t)h * Sz + s) * DHz;
  kdst[t] = (bf16)kvr[t];                                   // k_pass
  vT[((size_t)h * DVz + t) * Sz + s] = (bf16)kvr[DNz + t];  // V transposed
  if (t < DRz) {
    const float* rot = ckv + (size_t)s * (KLz + DRz) + KLz;
    const float c  = cosb[(size_t)s * DRz + t];
    const float sn = sinb[(size_t)s * DRz + t];
    float o;
    if (t < 32) o = rot[2 * t] * c - rot[2 * t + 1] * sn;
    else { int jj = t - 32; o = rot[2 * jj + 1] * c + rot[2 * jj] * sn; }
    kdst[DNz + t] = (bf16)o;
  }
}

// ---------------- flash attention, one 16-row q tile per wave --------------
__global__ __launch_bounds__(256) void attn_kernel(
    const bf16* __restrict__ qq, const bf16* __restrict__ kk,
    const bf16* __restrict__ vT, bf16* __restrict__ outh)
{
  __shared__ __align__(16) bf16 pstage[8][16 * 32];
  const int lane = threadIdx.x & 31;
  const int wave = threadIdx.x >> 5;
  const int h  = blockIdx.y;
  const int q0 = (blockIdx.x * 8 + wave) * 16;

  const bf16* Q  = qq + ((size_t)h * Sz + q0) * DHz;
  const bf16* Kh = kk + (size_t)h * Sz * DHz;
  const bf16* Vt = vT + (size_t)h * DVz * Sz;

  v16bf qf[6];
#pragma unroll
  for (int i = 0; i < 6; ++i) qf[i] = load_a_frag(Q + 32 * i, DHz, lane);

  v8f oacc[8];
#pragma unroll
  for (int j = 0; j < 8; ++j) oacc[j] = vzero8();
  float Mrow[8], Lrow[8];
#pragma unroll
  for (int v = 0; v < 8; ++v) { Mrow[v] = -1e30f; Lrow[v] = 0.f; }

  const int half = lane >> 4;
  const int col  = lane & 15;
  const int kend = q0 + 16;
  bf16* pb = &pstage[wave][0];

  for (int kb = 0; kb < kend; kb += 32) {
#pragma unroll
    for (int sb = 0; sb < 2; ++sb) {
      const int k16 = kb + sb * 16;
      v8f sc = vzero8();
      if (k16 < kend) {
        // group the 6 K-fragment loads so one wait covers the 6-WMMA chain
        v16bf kf[6];
#pragma unroll
        for (int i = 0; i < 6; ++i)
          kf[i] = load_b_frag(Kh + (size_t)k16 * DHz + 32 * i, DHz, lane);
#pragma unroll
        for (int i = 0; i < 6; ++i) sc = wmma_bf16(qf[i], kf[i], sc);
      }
      float alpha[8];
#pragma unroll
      for (int v = 0; v < 8; ++v) {
        const int qrow = q0 + v + 8 * half;
        const int kidx = k16 + col;
        float sv = (k16 < kend && kidx <= qrow) ? sc[v] * SCALING_F : -1e30f;
        float mx = sv;
#pragma unroll
        for (int off = 1; off < 16; off <<= 1) mx = fmaxf(mx, __shfl_xor(mx, off, 32));
        const float Mn = fmaxf(Mrow[v], mx);
        const float a  = __expf(Mrow[v] - Mn);
        const float p  = __expf(sv - Mn);
        float ps = p;
#pragma unroll
        for (int off = 1; off < 16; off <<= 1) ps += __shfl_xor(ps, off, 32);
        Lrow[v] = Lrow[v] * a + ps;
        Mrow[v] = Mn;
        alpha[v] = a;
        pb[(v + 8 * half) * 32 + sb * 16 + col] = (bf16)p;
      }
#pragma unroll
      for (int j = 0; j < 8; ++j)
#pragma unroll
        for (int v = 0; v < 8; ++v) oacc[j][v] *= alpha[v];
    }
    // issue V-fragment loads before the LDS round-trip so they overlap it
    v16bf vf[8];
#pragma unroll
    for (int j = 0; j < 8; ++j)
      vf[j] = load_b_frag(Vt + (size_t)(j * 16) * Sz + kb, Sz, lane);
    asm volatile("s_wait_dscnt 0" ::: "memory");   // p staged in LDS is visible
    v16bf pf = load_a_frag(pb, 32, lane);
#pragma unroll
    for (int j = 0; j < 8; ++j) oacc[j] = wmma_bf16(pf, vf[j], oacc[j]);
  }

#pragma unroll
  for (int j = 0; j < 8; ++j)
#pragma unroll
    for (int v = 0; v < 8; ++v) {
      const float o = oacc[j][v] / Lrow[v];
      outh[(size_t)(q0 + v + 8 * half) * (NHz * DVz) + h * DVz + j * 16 + col] = (bf16)o;
    }
}

// ---------------- host-side orchestration ----------------
extern "C" void kernel_launch(void* const* d_in, const int* in_sizes, int n_in,
                              void* d_out, int out_size, void* d_ws, size_t ws_size,
                              hipStream_t stream) {
  (void)in_sizes; (void)n_in; (void)out_size; (void)ws_size;
  const float* hidden   = (const float*)d_in[0];
  const float* cosb     = (const float*)d_in[1];
  const float* sinb     = (const float*)d_in[2];
  const float* w_qa     = (const float*)d_in[3];
  const float* w_qa_ln  = (const float*)d_in[4];
  const float* w_qb     = (const float*)d_in[5];
  const float* w_kva    = (const float*)d_in[6];
  const float* w_kva_ln = (const float*)d_in[7];
  const float* w_kvb    = (const float*)d_in[8];
  const float* w_o      = (const float*)d_in[9];

  char* ws = (char*)d_ws;
  size_t off = 0;
  auto alloc = [&](size_t bytes) -> void* {
    void* p = ws + off;
    off += (bytes + 255) & ~(size_t)255;
    return p;
  };

  bf16* hid_bf   = (bf16*)alloc((size_t)Sz * HIDz * 2);
  bf16* wqa_bf   = (bf16*)alloc((size_t)QLz * HIDz * 2);
  bf16* wqb_bf   = (bf16*)alloc((size_t)NHz * DHz * QLz * 2);
  bf16* wkva_bf  = (bf16*)alloc((size_t)(KLz + DRz) * HIDz * 2);
  bf16* wkvb_bf  = (bf16*)alloc((size_t)NHz * (DNz + DVz) * KLz * 2);
  bf16* wo_bf    = (bf16*)alloc((size_t)HIDz * NHz * DVz * 2);
  float* qlatraw = (float*)alloc((size_t)Sz * QLz * 4);
  bf16* qlat_bf  = (bf16*)alloc((size_t)Sz * QLz * 2);
  float* qraw    = (float*)alloc((size_t)Sz * NHz * DHz * 4);
  float* ckv     = (float*)alloc((size_t)Sz * (KLz + DRz) * 4);
  bf16* klat_bf  = (bf16*)alloc((size_t)Sz * KLz * 2);
  float* kvraw   = (float*)alloc((size_t)Sz * NHz * (DNz + DVz) * 4);
  bf16* qq       = (bf16*)alloc((size_t)NHz * Sz * DHz * 2);
  bf16* kk       = (bf16*)alloc((size_t)NHz * Sz * DHz * 2);
  bf16* vT       = (bf16*)alloc((size_t)NHz * DVz * Sz * 2);
  bf16* outh     = (bf16*)alloc((size_t)Sz * NHz * DVz * 2);

  auto cvt = [&](const float* x, bf16* y, int n) {
    f2bf_kernel<<<dim3((n + 255) / 256), dim3(256), 0, stream>>>(x, y, n);
  };
  cvt(hidden, hid_bf,  Sz * HIDz);
  cvt(w_qa,   wqa_bf,  QLz * HIDz);
  cvt(w_qb,   wqb_bf,  NHz * DHz * QLz);
  cvt(w_kva,  wkva_bf, (KLz + DRz) * HIDz);
  cvt(w_kvb,  wkvb_bf, NHz * (DNz + DVz) * KLz);
  cvt(w_o,    wo_bf,   HIDz * NHz * DVz);

  auto gemm = [&](const bf16* A, const bf16* W, float* C, int M, int N, int K) {
    dim3 g((N + 127) / 128, (M + 63) / 64);
    gemm_bt_kernel<<<g, dim3(256), 0, stream>>>(A, W, C, M, N, K);
  };

  // q path
  gemm(hid_bf, wqa_bf, qlatraw, Sz, QLz, HIDz);
  rmsnorm_kernel<<<dim3(Sz), dim3(256), 0, stream>>>(qlatraw, w_qa_ln, qlat_bf, QLz, QLz);
  gemm(qlat_bf, wqb_bf, qraw, Sz, NHz * DHz, QLz);
  build_q_kernel<<<dim3(Sz, NHz), dim3(192), 0, stream>>>(qraw, cosb, sinb, qq);

  // kv path
  gemm(hid_bf, wkva_bf, ckv, Sz, KLz + DRz, HIDz);
  rmsnorm_kernel<<<dim3(Sz), dim3(256), 0, stream>>>(ckv, w_kva_ln, klat_bf, KLz + DRz, KLz);
  gemm(klat_bf, wkvb_bf, kvraw, Sz, NHz * (DNz + DVz), KLz);
  build_k_kernel<<<dim3(Sz, NHz), dim3(128), 0, stream>>>(kvraw, ckv, cosb, sinb, kk, vT);

  // attention
  attn_kernel<<<dim3(Sz / 16 / 8, NHz), dim3(256), 0, stream>>>(qq, kk, vT, outh);

  // output projection -> f32 d_out
  gemm(outh, wo_bf, (float*)d_out, Sz, HIDz, NHz * DVz);
}
